// MaskedSelfAttention_71012989272337
// MI455X (gfx1250) — matrix operand order repaired and checked by reading, not compiled, not run on hardware
//
#include <hip/hip_runtime.h>
#include <hip/hip_bf16.h>

#define B_ 2
#define T_ 4096
#define C_ 512
#define H_ 8
#define D_ 64
#define BT_ (B_ * T_)
#define KSTRIDE 72          // LDS row stride in bf16 elems (144B, 16B-aligned, bank-skewed)
#define NEG_INF (-__builtin_inff())
#define SCALE 0.125f        // 1/sqrt(64)

typedef __attribute__((ext_vector_type(16))) __bf16 v16bf;
typedef __attribute__((ext_vector_type(8)))  float  v8f;

// GCC-style int4 vector: matches the V4i pointee type of the async builtins.
typedef int int4v __attribute__((vector_size(16)));
typedef __attribute__((address_space(1))) int4v g_int4;
typedef __attribute__((address_space(3))) int4v s_int4;

union Frag { v16bf v; uint4 q[2]; };

__device__ __forceinline__ v8f wmma_bf16(v16bf a, v16bf b, v8f c) {
  // D = A(16x32 bf16) * B(32x16 bf16) + C(16x16 f32)
  return __builtin_amdgcn_wmma_f32_16x16x32_bf16(false, a, false, b,
                                                 (short)0, c, false, false);
}

__device__ __forceinline__ v8f zero8() {
  v8f z;
#pragma unroll
  for (int i = 0; i < 8; ++i) z[i] = 0.f;
  return z;
}

// --- CDNA5 async global->LDS copy (ASYNCcnt path), with sync fallback -------
#if __has_builtin(__builtin_amdgcn_global_load_async_to_lds_b128)
#define HAS_ASYNC_LDS 1
#else
#define HAS_ASYNC_LDS 0
#endif

__device__ __forceinline__ void async_copy16(const __bf16* gsrc, __bf16* ldst) {
#if HAS_ASYNC_LDS
  __builtin_amdgcn_global_load_async_to_lds_b128(
      (g_int4*)gsrc, (s_int4*)ldst, 0, 0);
#else
  *(uint4*)ldst = *(const uint4*)gsrc;
#endif
}

__device__ __forceinline__ void wait_async() {
#if HAS_ASYNC_LDS
#if __has_builtin(__builtin_amdgcn_s_wait_asynccnt)
  __builtin_amdgcn_s_wait_asynccnt(0);
#else
  asm volatile("s_wait_asynccnt 0" ::: "memory");
#endif
#endif
}

// ---------------------------------------------------------------------------
// Kernel 0: one-shot f32 -> bf16 conversion of x and the three weight
// matrices (pure bandwidth; removes all cvt work from the GEMM hot loops).
// ---------------------------------------------------------------------------
__global__ __launch_bounds__(256)
void cvt_bf16_kernel(const float* __restrict__ x,
                     const float* __restrict__ Wq, const float* __restrict__ Wk,
                     const float* __restrict__ Wv,
                     __bf16* __restrict__ xb, __bf16* __restrict__ Wb) {
  const int z = blockIdx.y;          // 0: x, 1..3: Wq/Wk/Wv
  const float* src;
  __bf16* dst;
  size_t n;
  if (z == 0) { src = x; dst = xb; n = (size_t)BT_ * C_; }
  else {
    src = (z == 1) ? Wq : (z == 2) ? Wk : Wv;
    dst = Wb + (size_t)(z - 1) * C_ * C_;
    n   = (size_t)C_ * C_;
  }
  const size_t i = ((size_t)blockIdx.x * 256 + threadIdx.x) * 8;
  if (i >= n) return;
  float4 f0 = *(const float4*)(src + i);
  float4 f1 = *(const float4*)(src + i + 4);
  __align__(16) __bf16 h[8] = {(__bf16)f0.x, (__bf16)f0.y, (__bf16)f0.z, (__bf16)f0.w,
                               (__bf16)f1.x, (__bf16)f1.y, (__bf16)f1.z, (__bf16)f1.w};
  *(uint4*)(dst + i) = *(const uint4*)h;
}

// ---------------------------------------------------------------------------
// Kernel 1: QKV projection GEMM on bf16 inputs.  y = x @ W^T + b, emitted as
// bf16 [B,H,T,D].  Block = one 16-row M tile x full N=512 of one matrix;
// wave w owns head h=w (16x64 strip): A fragment loaded once per K-step and
// reused by 4 WMMAs -> 64 WMMAs per wave.
// ---------------------------------------------------------------------------
__global__ __launch_bounds__(256, 2)
void qkv_gemm_kernel(const __bf16* __restrict__ xb, const __bf16* __restrict__ Wb,
                     const float* __restrict__ bq, const float* __restrict__ bk,
                     const float* __restrict__ bv,
                     __bf16* __restrict__ Qb, __bf16* __restrict__ Kb,
                     __bf16* __restrict__ Vb) {
  const int tid  = threadIdx.x;
  const int wave = tid >> 5;
  const int lane = tid & 31;
  const int nl   = lane & 15;
  const int g    = lane >> 4;

  const int mt = blockIdx.x;       // 0..511 (M/16 tiles, M = B*T)
  const int z  = blockIdx.y;       // 0:Q 1:K 2:V

  const float* bias = (z == 0) ? bq : (z == 1) ? bk : bv;
  __bf16*      Yb   = (z == 0) ? Qb : (z == 1) ? Kb : Vb;

  const int m0 = mt * 16;
  const int n0 = wave * 64;        // wave == head
  const uint4* xp = (const uint4*)(xb + (size_t)(m0 + nl) * C_);
  const uint4* wp[4];
#pragma unroll
  for (int jd = 0; jd < 4; ++jd)
    wp[jd] = (const uint4*)(Wb + ((size_t)z * C_ + n0 + jd * 16 + nl) * C_);

  v8f acc[4];
#pragma unroll
  for (int jd = 0; jd < 4; ++jd) acc[jd] = zero8();

  for (int k0 = 0; k0 < C_; k0 += 32) {
    const int k8 = k0 >> 3;                // uint4 index of K base
    Frag a;
    a.q[0] = xp[k8 + g];                   // K = k0 + g*8 + e
    a.q[1] = xp[k8 + 2 + g];               // K = k0 + 16 + g*8 + e
#pragma unroll
    for (int jd = 0; jd < 4; ++jd) {
      Frag b;
      b.q[0] = wp[jd][k8 + g];             // K = k0 + g*8 + e
      b.q[1] = wp[jd][k8 + 2 + g];
      acc[jd] = wmma_bf16(a.v, b.v, acc[jd]);
    }
  }

  const int h = wave;
#pragma unroll
  for (int jd = 0; jd < 4; ++jd) {
    const float bb = bias[n0 + jd * 16 + nl];
#pragma unroll
    for (int r = 0; r < 8; ++r) {
      const int m  = m0 + r + 8 * g;       // C/D layout: VGPR r -> M = r + 8*(lane/16)
      const int bi = m >> 12;              // m / T
      const int t  = m & (T_ - 1);
      Yb[(((size_t)bi * H_ + h) * T_ + t) * D_ + jd * 16 + nl] =
          (__bf16)(acc[jd][r] + bb);
    }
  }
}

// ---------------------------------------------------------------------------
// Kernel 2: causal flash attention, one (b,h) x 128-query tile per block.
// 8 waves; wave owns 16 queries. Online softmax in f32, matmuls in bf16 WMMA.
// K tiles double-buffered via async global->LDS; V fetch overlapped with
// compute, transposed into LDS after the WMMA work.
// ---------------------------------------------------------------------------
__global__ __launch_bounds__(256, 1)
void attn_kernel(const __bf16* __restrict__ Qb, const __bf16* __restrict__ Kb,
                 const __bf16* __restrict__ Vb, float* __restrict__ out) {
  __shared__ __align__(16) __bf16 Ksh[2][64 * KSTRIDE];     // [key][d]
  __shared__ __align__(16) __bf16 Vsh[2][64 * KSTRIDE];     // transposed: [d][key]
  __shared__ __align__(16) __bf16 Psh[8 * 16 * KSTRIDE];    // per-wave P bounce

  const int tid  = threadIdx.x;
  const int wave = tid >> 5, lane = tid & 31;
  const int nl = lane & 15, g = lane >> 4;

  const int bh   = blockIdx.y;                 // 0..15
  const int qblk = blockIdx.x;                 // 0..31
  const int q0   = qblk * 128 + wave * 16;     // wave's first query
  const size_t base = (size_t)bh * T_ * D_;

  // cooperative-fill geometry (per thread)
  const int frow = tid >> 3;            // 0..31
  const int fcol = (tid & 7) * 8;       // 0..56

  auto issueK = [&](int kb, int buf) {
#pragma unroll
    for (int c = 0; c < 2; ++c) {       // 2 x 16B chunks per thread = 8KB tile
      const int row = frow + c * 32;
      async_copy16(Kb + base + (size_t)(kb + row) * D_ + fcol,
                   &Ksh[buf][row * KSTRIDE + fcol]);
    }
  };
  auto loadV = [&](int kb, uint4& v0, uint4& v1) {
    const int row = frow * 2;
    v0 = *(const uint4*)(Vb + base + (size_t)(kb + row) * D_ + fcol);
    v1 = *(const uint4*)(Vb + base + (size_t)(kb + row + 1) * D_ + fcol);
  };
  auto storeVT = [&](int buf, uint4 v0, uint4 v1) {
    const int row = frow * 2;
    const unsigned short* h0 = (const unsigned short*)&v0;
    const unsigned short* h1 = (const unsigned short*)&v1;
#pragma unroll
    for (int j = 0; j < 8; ++j) {       // packed {k,k+1} per d -> b32 stores
      const unsigned int pk = (unsigned int)h0[j] | ((unsigned int)h1[j] << 16);
      *(unsigned int*)(&Vsh[buf][(fcol + j) * KSTRIDE + row]) = pk;
    }
  };

  // Persistent Q A-fragments (16 queries x D=64 -> two 16x32 fragments)
  Frag aq0, aq1;
  {
    const uint4* qp = (const uint4*)(Qb + base + (size_t)(q0 + nl) * D_);
    aq0.q[0] = qp[g];     aq0.q[1] = qp[2 + g];
    aq1.q[0] = qp[4 + g]; aq1.q[1] = qp[6 + g];
  }

  v8f o[4];
#pragma unroll
  for (int j = 0; j < 4; ++j) o[j] = zero8();
  float mrun[8], lrun[8];
#pragma unroll
  for (int r = 0; r < 8; ++r) { mrun[r] = NEG_INF; lrun[r] = 0.f; }

  const int nkb = qblk * 2 + 2;   // 64-key blocks needed to cover causal range

  // prologue: stage tile 0 into buffer 0
  {
    issueK(0, 0);
    uint4 v0, v1;
    loadV(0, v0, v1);
    storeVT(0, v0, v1);
    wait_async();
    __syncthreads();
  }

  for (int kb64 = 0; kb64 < nkb; ++kb64) {
    const int kb = kb64 * 64;
    const int p  = kb64 & 1;
    const bool more = (kb64 + 1 < nkb);

    uint4 nv0 = {0, 0, 0, 0}, nv1 = {0, 0, 0, 0};
    if (more) {                         // overlap next tile fetch with compute
      issueK(kb + 64, p ^ 1);
      loadV(kb + 64, nv0, nv1);
    }

    if (kb <= q0 + 15) {                // causal: fully-masked waves skip compute
      // ---- S = Q K^T (4 key subtiles of 16, 2 WMMAs each over D=64) ----
      v8f s[4];
#pragma unroll
      for (int j = 0; j < 4; ++j) {
        s[j] = zero8();
        const uint4* kp = (const uint4*)(&Ksh[p][(j * 16 + nl) * KSTRIDE]);
        Frag b0, b1;
        b0.q[0] = kp[g];     b0.q[1] = kp[2 + g];      // d = g*8 + e, 16 + g*8 + e
        b1.q[0] = kp[4 + g]; b1.q[1] = kp[6 + g];      // d = 32 + g*8 + e, 48 + g*8 + e
        s[j] = wmma_bf16(aq0.v, b0.v, s[j]);
        s[j] = wmma_bf16(aq1.v, b1.v, s[j]);
      }

      // ---- online softmax (f32). Row m = r + 8*g lives in a 16-lane half ----
      float alpha[8];
#pragma unroll
      for (int r = 0; r < 8; ++r) {
        const int qg = q0 + r + 8 * g;
        float mx = NEG_INF;
#pragma unroll
        for (int j = 0; j < 4; ++j) {
          const int kg = kb + j * 16 + nl;
          float vv = s[j][r] * SCALE;
          vv = (kg > qg) ? NEG_INF : vv;
          s[j][r] = vv;
          mx = fmaxf(mx, vv);
        }
#pragma unroll
        for (int msk = 1; msk < 16; msk <<= 1)
          mx = fmaxf(mx, __shfl_xor(mx, msk, 32));
        const float mo = mrun[r];
        const float mn = fmaxf(mo, mx);
        const float a  = (mo == NEG_INF) ? 0.f : __expf(mo - mn);
        float rs = 0.f;
#pragma unroll
        for (int j = 0; j < 4; ++j) {
          const float pp = __expf(s[j][r] - mn);
          s[j][r] = pp;
          rs += pp;
        }
#pragma unroll
        for (int msk = 1; msk < 16; msk <<= 1) rs += __shfl_xor(rs, msk, 32);
        lrun[r] = lrun[r] * a + rs;
        mrun[r] = mn;
        alpha[r] = a;
      }
#pragma unroll
      for (int j = 0; j < 4; ++j)
#pragma unroll
        for (int r = 0; r < 8; ++r) o[j][r] *= alpha[r];

      // ---- bounce P through per-wave LDS: C-layout -> A-layout (bf16) ----
      __bf16* pw = &Psh[wave * 16 * KSTRIDE];
#pragma unroll
      for (int j = 0; j < 4; ++j)
#pragma unroll
        for (int r = 0; r < 8; ++r)
          pw[(r + 8 * g) * KSTRIDE + j * 16 + nl] = (__bf16)s[j][r];

      Frag pa0, pa1;
      {
        const uint4* pp = (const uint4*)(pw + nl * KSTRIDE);
        pa0.q[0] = pp[g];     pa0.q[1] = pp[2 + g];
        pa1.q[0] = pp[4 + g]; pa1.q[1] = pp[6 + g];
      }

      // ---- O += P V (contraction over 64 keys; V^T staged in LDS) ----
#pragma unroll
      for (int jd = 0; jd < 4; ++jd) {
        const uint4* vp = (const uint4*)(&Vsh[p][(jd * 16 + nl) * KSTRIDE]);
        Frag b0, b1;
        b0.q[0] = vp[g];     b0.q[1] = vp[2 + g];
        b1.q[0] = vp[4 + g]; b1.q[1] = vp[6 + g];
        o[jd] = wmma_bf16(pa0.v, b0.v, o[jd]);
        o[jd] = wmma_bf16(pa1.v, b1.v, o[jd]);
      }
    }

    if (more) storeVT(p ^ 1, nv0, nv1);   // V transpose after compute
    wait_async();                         // K tile p^1 resident in LDS
    __syncthreads();
  }

  // ---- epilogue: O / l, back to [B,T,C] f32 ----
  const int b = bh >> 3, h = bh & 7;
#pragma unroll
  for (int jd = 0; jd < 4; ++jd)
#pragma unroll
    for (int r = 0; r < 8; ++r) {
      const int qg = q0 + r + 8 * g;
      out[((size_t)b * T_ + qg) * C_ + h * D_ + jd * 16 + nl] =
          o[jd][r] * (1.f / lrun[r]);
    }
}

// ---------------------------------------------------------------------------
extern "C" void kernel_launch(void* const* d_in, const int* in_sizes, int n_in,
                              void* d_out, int out_size, void* d_ws, size_t ws_size,
                              hipStream_t stream) {
  const float* x  = (const float*)d_in[0];
  const float* Wq = (const float*)d_in[1];
  const float* bq = (const float*)d_in[2];
  const float* Wk = (const float*)d_in[3];
  const float* bk = (const float*)d_in[4];
  const float* Wv = (const float*)d_in[5];
  const float* bv = (const float*)d_in[6];
  float* out = (float*)d_out;

  const size_t perBuf = (size_t)B_ * H_ * T_ * D_;   // 4 Mi bf16 = 8 MB each
  __bf16* Qb = (__bf16*)d_ws;
  __bf16* Kb = Qb + perBuf;
  __bf16* Vb = Kb + perBuf;
  __bf16* xb = Vb + perBuf;                          // [B*T, C] bf16
  __bf16* Wb = xb + (size_t)BT_ * C_;                // [3, C, C] bf16

  // 0) f32 -> bf16 staging
  dim3 g0((BT_ * C_) / (256 * 8), 4);                // 2048 x 4 (W rows guarded)
  cvt_bf16_kernel<<<g0, 256, 0, stream>>>(x, Wq, Wk, Wv, xb, Wb);

  // 1) QKV projections
  dim3 g1(BT_ / 16, 3);                              // 512 x 3
  qkv_gemm_kernel<<<g1, 256, 0, stream>>>(xb, Wb, bq, bk, bv, Qb, Kb, Vb);

  // 2) causal flash attention
  dim3 g2(T_ / 128, B_ * H_);                        // 32 x 16
  attn_kernel<<<g2, 256, 0, stream>>>(Qb, Kb, Vb, out);
}
